// AllegroBond_81578608820655
// MI455X (gfx1250) — matrix operand order repaired and checked by reading, not compiled, not run on hardware
//
#include <hip/hip_runtime.h>
#include <hip/hip_bf16.h>

typedef __attribute__((ext_vector_type(16))) _Float16 v16h;
typedef __attribute__((ext_vector_type(8)))  _Float16 v8h;
typedef __attribute__((ext_vector_type(8)))  float    v8f;
typedef int v4i __attribute__((vector_size(16)));
typedef __attribute__((address_space(1))) v4i* as1_v4i;
typedef __attribute__((address_space(3))) v4i* as3_v4i;

static constexpr int   EDGES = 800000;
static constexpr int   NODES = 50000;
static constexpr float LN2   = 0.69314718056f;

#if __has_builtin(__builtin_amdgcn_global_load_async_to_lds_b128)
#define HAVE_ASYNC_LDS 1
#else
#define HAVE_ASYNC_LDS 0
#endif

// ---------------- helpers ----------------

// branchless ssp(x) = softplus(x) - ln2 = max(x,0) + log(1+exp(-|x|)) - ln2
__device__ __forceinline__ float ssp_f(float x) {
  float ax = __builtin_fabsf(x);
  float t  = __expf(-ax);
  return fmaxf(x, 0.0f) + __logf(1.0f + t) - LN2;
}

// 16-byte global -> LDS copy; async DMA (GLOBAL_LOAD_ASYNC_TO_LDS_B128) when available
__device__ __forceinline__ void copy16_g2l(void* lds_dst, const void* gsrc) {
#if HAVE_ASYNC_LDS
  __builtin_amdgcn_global_load_async_to_lds_b128(
      (as1_v4i)(v4i*)const_cast<void*>(gsrc),
      (as3_v4i)(v4i*)lds_dst, 0, 0);
#else
  *(uint4*)lds_dst = *(const uint4*)gsrc;
#endif
}

__device__ __forceinline__ void wait_async_copies() {
#if HAVE_ASYNC_LDS
#if __has_builtin(__builtin_amdgcn_s_wait_asynccnt)
  __builtin_amdgcn_s_wait_asynccnt(0);
#else
  asm volatile("s_wait_asynccnt 0" ::: "memory");
#endif
#endif
}

// A fragment (16xK tile, row-major, stride ld halves). ISA 16-bit A 16x32 layout:
// lane L holds row mbase+(L&15); element e <-> k = kb + (e&7) + 8*(L>>4) + 16*(e>>3)
__device__ __forceinline__ v16h frag_a(const _Float16* X, int ld, int mbase, int lane, int kb) {
  int row = mbase + (lane & 15);
  int h   = lane >> 4;
  const _Float16* p = X + (long)row * ld + kb + 8 * h;
  v8h lo = *(const v8h*)(p);
  v8h hi = *(const v8h*)(p + 16);
  v16h a;
#pragma unroll
  for (int i = 0; i < 8; i++) { a[i] = lo[i]; a[i + 8] = hi[i]; }
  return a;
}

// B fragment from transposed weights Wt[n][k] (stride ld halves). ISA 16-bit B 32x16:
// lane L holds column n = nbase+(L&15); element e <-> k = kb + (e&7) + 8*(L>>4) + 16*(e>>3)
__device__ __forceinline__ v16h frag_b(const _Float16* Wt, int ld, int nbase, int lane, int kb) {
  int n  = nbase + (lane & 15);
  int k0 = kb + ((lane >> 4) << 3);
  const _Float16* p = Wt + (long)n * ld + k0;
  v8h lo = *(const v8h*)(p);
  v8h hi = *(const v8h*)(p + 16);
  v16h b;
#pragma unroll
  for (int i = 0; i < 8; i++) { b[i] = lo[i]; b[i + 8] = hi[i]; }
  return b;
}

__device__ __forceinline__ v8f wmma32(v16h a, v16h b, v8f c) {
  return __builtin_amdgcn_wmma_f32_16x16x32_f16(false, a, false, b, (short)0, c, false, false);
}

// ---------------- weight prep: f32 [fin][fout] -> f16 transposed [fout][kpad] ----------------

__global__ __launch_bounds__(256) void k_prep_w(const float* __restrict__ W,
                                                _Float16* __restrict__ Wt,
                                                int fin, int fout, int kpad) {
  int idx = blockIdx.x * 256 + threadIdx.x;
  if (idx >= fout * kpad) return;
  int n = idx / kpad, k = idx % kpad;
  Wt[idx] = (k < fin) ? (_Float16)W[k * fout + n] : (_Float16)0.0f;
}

// ---------------- featurize: build 96-wide padded f16 input + uij ----------------

__global__ __launch_bounds__(256) void k_feat(const int* __restrict__ species,
                                              const int* __restrict__ esrc,
                                              const int* __restrict__ edst,
                                              const int* __restrict__ bo,
                                              const float* __restrict__ dist,
                                              const float* __restrict__ vecp,
                                              const float* __restrict__ ecfp,
                                              _Float16* __restrict__ xin,
                                              float* __restrict__ uij) {
  long e = (long)blockIdx.x * 256 + threadIdx.x;
  if (e >= EDGES) return;
  _Float16* p = xin + e * 96;
#pragma unroll
  for (int i = 0; i < 96; i++) p[i] = (_Float16)0.0f;
  int s = esrc[e], d = edst[e];
  p[species[s]]      = (_Float16)1.0f;
  p[16 + species[d]] = (_Float16)1.0f;
  float r = dist[e];
  const float sq = 0.632455532f;  // sqrt(2/5)
#pragma unroll
  for (int k = 0; k < 8; k++)
    p[32 + k] = (_Float16)(sq * __sinf((float)(k + 1) * 0.6283185307f * r) / r);
  const float wbo[6] = {1.0f, 1.5f, 2.0f, 0.5f, 3.0f, 0.25f};
  p[40] = (_Float16)wbo[bo[e]];
#pragma unroll
  for (int j = 0; j < 16; j++) p[41 + j] = (_Float16)ecfp[(long)s * 16 + j];
#pragma unroll
  for (int j = 0; j < 16; j++) p[57 + j] = (_Float16)ecfp[(long)d * 16 + j];
  uij[e * 4 + 0] = 1.0f;
#pragma unroll
  for (int i = 0; i < 3; i++) uij[e * 4 + 1 + i] = vecp[e * 3 + i] / r;
}

// ---------------- fused 3-layer MLP (96->64->64->64, ssp, *sw, optional residual) ----------------
// 64 edges / block, 8 waves; each wave: one 16-wide n-tile, two 16-tall m-tiles (B reuse).

__global__ __launch_bounds__(256) void k_mlp(const _Float16* __restrict__ xin,   // E x 96
                                             const _Float16* __restrict__ netW,  // 14336 halves
                                             const float* __restrict__ b0,
                                             const float* __restrict__ b1,
                                             const float* __restrict__ b2,
                                             const float* __restrict__ sw,
                                             float* __restrict__ x,
                                             _Float16* __restrict__ xh,
                                             int residual) {
  __shared__ __align__(16) _Float16 sW[14336];   // Wt0[64][96] | Wt1[64][64] | Wt2[64][64]
  __shared__ __align__(16) _Float16 sX[64 * 96];
  __shared__ __align__(16) _Float16 sHa[64 * 64];
  __shared__ __align__(16) _Float16 sHb[64 * 64];
  __shared__ float sB[192];

  int t = threadIdx.x;
  long e0 = (long)blockIdx.x * 64;
  { // tile fills: async DMA to LDS when available
    const char* wsrc = (const char*)netW;
    char* wdst = (char*)sW;
#pragma unroll 2
    for (int i = t; i < 1792; i += 256) copy16_g2l(wdst + i * 16, wsrc + i * 16);
    const char* xsrc = (const char*)(xin + e0 * 96);
    char* xdst = (char*)sX;
#pragma unroll 2
    for (int i = t; i < 768; i += 256) copy16_g2l(xdst + i * 16, xsrc + i * 16);
    if (t < 64) { sB[t] = b0[t]; sB[64 + t] = b1[t]; sB[128 + t] = b2[t]; }
    wait_async_copies();
  }
  __syncthreads();

  int lane = t & 31, wid = t >> 5;
  int nt  = (wid & 3) * 16;          // 0,16,32,48
  int mt0 = (wid >> 2) * 16;         // 0 or 16
  int mt1 = mt0 + 32;                // 32 or 48
  int n   = nt + (lane & 15);
  int mh  = (lane >> 4) << 3;

  // layer 1: K=96
  v8f acc0 = {}, acc1 = {};
#pragma unroll
  for (int kb = 0; kb < 96; kb += 32) {
    v16h b = frag_b(sW, 96, nt, lane, kb);
    acc0 = wmma32(frag_a(sX, 96, mt0, lane, kb), b, acc0);
    acc1 = wmma32(frag_a(sX, 96, mt1, lane, kb), b, acc1);
  }
  {
    float bias = sB[n];
#pragma unroll
    for (int r = 0; r < 8; r++) {
      sHa[(mt0 + mh + r) * 64 + n] = (_Float16)ssp_f(acc0[r] + bias);
      sHa[(mt1 + mh + r) * 64 + n] = (_Float16)ssp_f(acc1[r] + bias);
    }
  }
  __syncthreads();

  // layer 2: K=64
  acc0 = (v8f){}; acc1 = (v8f){};
#pragma unroll
  for (int kb = 0; kb < 64; kb += 32) {
    v16h b = frag_b(sW + 6144, 64, nt, lane, kb);
    acc0 = wmma32(frag_a(sHa, 64, mt0, lane, kb), b, acc0);
    acc1 = wmma32(frag_a(sHa, 64, mt1, lane, kb), b, acc1);
  }
  {
    float bias = sB[64 + n];
#pragma unroll
    for (int r = 0; r < 8; r++) {
      sHb[(mt0 + mh + r) * 64 + n] = (_Float16)ssp_f(acc0[r] + bias);
      sHb[(mt1 + mh + r) * 64 + n] = (_Float16)ssp_f(acc1[r] + bias);
    }
  }
  __syncthreads();

  // layer 3: K=64, linear, *sw, (+x)
  acc0 = (v8f){}; acc1 = (v8f){};
#pragma unroll
  for (int kb = 0; kb < 64; kb += 32) {
    v16h b = frag_b(sW + 10240, 64, nt, lane, kb);
    acc0 = wmma32(frag_a(sHb, 64, mt0, lane, kb), b, acc0);
    acc1 = wmma32(frag_a(sHb, 64, mt1, lane, kb), b, acc1);
  }
  {
    float bias = sB[128 + n];
#pragma unroll
    for (int r = 0; r < 8; r++) {
      long ea = e0 + mt0 + mh + r;
      float va = (acc0[r] + bias) * sw[ea];
      float xa = residual ? (x[ea * 64 + n] + va) : va;
      x[ea * 64 + n]  = xa;
      xh[ea * 64 + n] = (_Float16)xa;
      long eb = e0 + mt1 + mh + r;
      float vb = (acc1[r] + bias) * sw[eb];
      float xb = residual ? (x[eb * 64 + n] + vb) : vb;
      x[eb * 64 + n]  = xb;
      xh[eb * 64 + n] = (_Float16)xb;
    }
  }
}

// ---------------- 64->16 projection: y = xh @ Wt16 (E x 16, f32) ----------------

__global__ __launch_bounds__(256) void k_gemm16(const _Float16* __restrict__ xh,
                                                const _Float16* __restrict__ Wt16, // [16][64]
                                                float* __restrict__ y) {
  __shared__ __align__(16) _Float16 sW[1024];
  int t = threadIdx.x;
  {
    const uint4* ws = (const uint4*)Wt16;
    uint4* wd = (uint4*)sW;
    if (t < 128) wd[t] = ws[t];
  }
  __syncthreads();
  int lane = t & 31, wid = t >> 5;
  long e0 = (long)blockIdx.x * 128 + wid * 16;
  v8f acc = {};
#pragma unroll
  for (int kb = 0; kb < 64; kb += 32)
    acc = wmma32(frag_a(xh + e0 * 64, 64, 0, lane, kb), frag_b(sW, 64, 0, lane, kb), acc);
  int n  = lane & 15;
  int mh = (lane >> 4) << 3;
#pragma unroll
  for (int r = 0; r < 8; r++) y[(e0 + mh + r) * 16 + n] = acc[r];
}

// ---------------- v init: v[e][a][c] = xwv[e][c] * uij[e][a] ----------------

__global__ __launch_bounds__(256) void k_outer(const float* __restrict__ xwv,
                                               const float* __restrict__ uij,
                                               float* __restrict__ v) {
  long idx = (long)blockIdx.x * 256 + threadIdx.x;
  long e = idx >> 6;
  int a = (int)((idx >> 4) & 3), c = (int)(idx & 15);
  v[idx] = xwv[e * 16 + c] * uij[e * 4 + a];
}

// ---------------- zero node accumulator ----------------

__global__ __launch_bounds__(256) void k_zero(float4* __restrict__ p) {
  long i = (long)blockIdx.x * 256 + threadIdx.x;
  p[i] = make_float4(0.f, 0.f, 0.f, 0.f);
}

// ---------------- segment-sum scatter: winode[src] += xw ⊗ uij ----------------

__global__ __launch_bounds__(256) void k_scatter(const int* __restrict__ esrc,
                                                 const float* __restrict__ xw,
                                                 const float* __restrict__ uij,
                                                 float* __restrict__ winode) {
  long idx = (long)blockIdx.x * 256 + threadIdx.x;  // E*16 threads
  long e = idx >> 4;
  int c = (int)(idx & 15);
  int s = esrc[e];
  float val = xw[idx];
  float* b = winode + (long)s * 64 + c;
  const float* u = uij + e * 4;
#pragma unroll
  for (int a = 0; a < 4; a++)
    __hip_atomic_fetch_add(b + a * 16, val * u[a], __ATOMIC_RELAXED, __HIP_MEMORY_SCOPE_AGENT);
}

// ---------------- combine: scalars, next-MLP input, v update ----------------

__global__ __launch_bounds__(256) void k_combine(const int* __restrict__ esrc,
                                                 const float* __restrict__ winode,
                                                 float* __restrict__ v,
                                                 const _Float16* __restrict__ xh,
                                                 float* __restrict__ s1o,
                                                 float* __restrict__ s2o,
                                                 _Float16* __restrict__ xin,
                                                 const float* __restrict__ Wvec,
                                                 const float* __restrict__ WscW,
                                                 const float* __restrict__ Wscb,
                                                 int do_vupd) {
  __shared__ float sWvec[768];
  __shared__ float sWsc[512];
  __shared__ float sb[16];
  int t = threadIdx.x;
  if (do_vupd) {
    for (int i = t; i < 768; i += 256) sWvec[i] = Wvec[i];
    for (int i = t; i < 512; i += 256) sWsc[i] = WscW[i];
    if (t < 16) sb[t] = Wscb[t];
  }
  __syncthreads();

  long e = (long)blockIdx.x * 256 + t;
  if (e >= EDGES) return;
  int s = esrc[e];
  float wi[64], vv[64];
  const float* wp = winode + (long)s * 64;
  float* vp = v + e * 64;
#pragma unroll
  for (int i = 0; i < 64; i++) wi[i] = wp[i];
#pragma unroll
  for (int i = 0; i < 64; i++) vv[i] = vp[i];

  float sc1[16], sc2[16];
#pragma unroll
  for (int c = 0; c < 16; c++) {
    sc1[c] = wi[c] * vv[c];
    sc2[c] = wi[16 + c] * vv[16 + c] + wi[32 + c] * vv[32 + c] + wi[48 + c] * vv[48 + c];
  }
#pragma unroll
  for (int c = 0; c < 16; c++) { s1o[e * 16 + c] = sc1[c]; s2o[e * 16 + c] = sc2[c]; }

  // next MLP input: [xh | sc1 | sc2]
  _Float16* xp = xin + e * 96;
#pragma unroll
  for (int i = 0; i < 64; i++) xp[i] = xh[e * 64 + i];
#pragma unroll
  for (int c = 0; c < 16; c++) { xp[64 + c] = (_Float16)sc1[c]; xp[80 + c] = (_Float16)sc2[c]; }

  if (do_vupd) {
    float acc0[16];
#pragma unroll
    for (int j = 0; j < 16; j++) acc0[j] = sb[j];
    for (int k = 0; k < 16; k++) {
      float a1 = sc1[k], a2 = sc2[k];
#pragma unroll
      for (int j = 0; j < 16; j++)
        acc0[j] += a1 * sWsc[k * 16 + j] + a2 * sWsc[(16 + k) * 16 + j];
    }
    float accv[48];
#pragma unroll
    for (int j = 0; j < 48; j++) accv[j] = 0.0f;
    for (int k = 0; k < 16; k++) {
      float w0 = wi[k], v0 = vv[k];
      float w1 = wi[16 + k], w2 = wi[32 + k], w3 = wi[48 + k];
      float u1 = vv[16 + k], u2 = vv[32 + k], u3 = vv[48 + k];
      float c1 = w2 * u3 - w3 * u2;   // cross rows (axis=1)
      float c2 = w3 * u1 - w1 * u3;
      float c3 = w1 * u2 - w2 * u1;
#pragma unroll
      for (int j = 0; j < 16; j++) {
        float wv1 = sWvec[k * 16 + j], wv2 = sWvec[(16 + k) * 16 + j], wv3 = sWvec[(32 + k) * 16 + j];
        accv[j]      += c1 * wv1 + (w0 * u1) * wv2 + (w1 * v0) * wv3;
        accv[16 + j] += c2 * wv1 + (w0 * u2) * wv2 + (w2 * v0) * wv3;
        accv[32 + j] += c3 * wv1 + (w0 * u3) * wv2 + (w3 * v0) * wv3;
      }
    }
#pragma unroll
    for (int j = 0; j < 16; j++) vp[j] = vv[j] + acc0[j];
#pragma unroll
    for (int j = 0; j < 48; j++) vp[16 + j] = vv[16 + j] + accv[j];
  }
}

// ---------------- host ----------------

extern "C" void kernel_launch(void* const* d_in, const int* in_sizes, int n_in,
                              void* d_out, int out_size, void* d_ws, size_t ws_size,
                              hipStream_t stream) {
  const int*   species = (const int*)d_in[0];
  const int*   esrc    = (const int*)d_in[1];
  const int*   edst    = (const int*)d_in[2];
  const int*   bo      = (const int*)d_in[3];
  const float* sw      = (const float*)d_in[4];
  const float* dist    = (const float*)d_in[5];
  const float* vec     = (const float*)d_in[6];
  const float* ecfp    = (const float*)d_in[7];
  // params flattened: Wsc[0].{W,b}, Wsc[1].{W,b}, Wv, Wvec[0..1], Ww[0..2], fc[0..3].{W0,W1,W2,b0,b1,b2}
  const float* WscW[2] = {(const float*)d_in[8], (const float*)d_in[10]};
  const float* Wscb[2] = {(const float*)d_in[9], (const float*)d_in[11]};
  const float* Wv      = (const float*)d_in[12];
  const float* Wvc[2]  = {(const float*)d_in[13], (const float*)d_in[14]};
  const float* Ww[3]   = {(const float*)d_in[15], (const float*)d_in[16], (const float*)d_in[17]};
  const float* fcW[4][3];
  const float* fcb[4][3];
  for (int j = 0; j < 3; j++) { fcW[0][j] = (const float*)d_in[18 + j]; fcb[0][j] = (const float*)d_in[21 + j]; }
  for (int l = 1; l < 4; l++)
    for (int j = 0; j < 3; j++) {
      fcW[l][j] = (const float*)d_in[24 + 6 * (l - 1) + j];
      fcb[l][j] = (const float*)d_in[27 + 6 * (l - 1) + j];
    }

  char* wsb = (char*)d_ws;
  size_t off = 0;
  auto alloc = [&](size_t b) { size_t o = off; off = (off + b + 255) & ~(size_t)255; return o; };
  _Float16* wt  = (_Float16*)(wsb + alloc((size_t)61440 * 2));
  _Float16* xh  = (_Float16*)(wsb + alloc((size_t)EDGES * 64 * 2));
  _Float16* xin = (_Float16*)(wsb + alloc((size_t)EDGES * 96 * 2));
  float* uij    = (float*)(wsb + alloc((size_t)EDGES * 4 * 4));
  float* v      = (float*)(wsb + alloc((size_t)EDGES * 64 * 4));
  float* xw     = (float*)(wsb + alloc((size_t)EDGES * 16 * 4));
  float* s1     = (float*)(wsb + alloc((size_t)EDGES * 16 * 4));
  float* s2     = (float*)(wsb + alloc((size_t)EDGES * 16 * 4));
  float* winode = (float*)(wsb + alloc((size_t)NODES * 64 * 4));
  float* x      = (float*)d_out;  // E x 64 fp32 output doubles as x buffer

  // --- weight prep (f16, transposed, K-padded) ---
  for (int l = 0; l < 4; l++) {
    int fin0 = (l == 0) ? 73 : 96;
    k_prep_w<<<(64 * 96 + 255) / 256, 256, 0, stream>>>(fcW[l][0], wt + l * 14336, fin0, 64, 96);
    k_prep_w<<<(64 * 64 + 255) / 256, 256, 0, stream>>>(fcW[l][1], wt + l * 14336 + 6144, 64, 64, 64);
    k_prep_w<<<(64 * 64 + 255) / 256, 256, 0, stream>>>(fcW[l][2], wt + l * 14336 + 10240, 64, 64, 64);
  }
  _Float16* wtWv = wt + 57344;
  _Float16* wtWw = wt + 58368;
  k_prep_w<<<(16 * 64 + 255) / 256, 256, 0, stream>>>(Wv, wtWv, 64, 16, 64);
  for (int l = 0; l < 3; l++)
    k_prep_w<<<(16 * 64 + 255) / 256, 256, 0, stream>>>(Ww[l], wtWw + l * 1024, 64, 16, 64);

  // --- pipeline ---
  k_feat<<<EDGES / 256, 256, 0, stream>>>(species, esrc, edst, bo, dist, vec, ecfp, xin, uij);
  k_mlp<<<EDGES / 64, 256, 0, stream>>>(xin, wt, fcb[0][0], fcb[0][1], fcb[0][2], sw, x, xh, 0);
  k_gemm16<<<EDGES / 128, 256, 0, stream>>>(xh, wtWv, xw);
  k_outer<<<(EDGES * 64) / 256, 256, 0, stream>>>(xw, uij, v);

  for (int l = 0; l < 3; l++) {
    k_gemm16<<<EDGES / 128, 256, 0, stream>>>(xh, wtWw + l * 1024, xw);
    k_zero<<<(NODES * 64 / 4) / 256, 256, 0, stream>>>((float4*)winode);
    k_scatter<<<(EDGES * 16) / 256, 256, 0, stream>>>(esrc, xw, uij, winode);
    int dv = (l < 2) ? 1 : 0;
    k_combine<<<EDGES / 256, 256, 0, stream>>>(esrc, winode, v, xh, s1, s2, xin,
                                               Wvc[dv ? l : 0], WscW[dv ? l : 0],
                                               Wscb[dv ? l : 0], dv);
    k_mlp<<<EDGES / 64, 256, 0, stream>>>(xin, wt + (l + 1) * 14336,
                                          fcb[l + 1][0], fcb[l + 1][1], fcb[l + 1][2],
                                          sw, x, xh, 1);
  }
  (void)in_sizes; (void)n_in; (void)out_size; (void)ws_size;
}
